// BSNLayer_42348377538688
// MI455X (gfx1250) — matrix operand order repaired and checked by reading, not compiled, not run on hardware
//
#include <hip/hip_runtime.h>
#include <cstdint>
#include <cstddef>

// Problem shape (fixed by the reference): x (16, 512, 64, 64), v (16, 512, 1)
#define B_DIM 16
#define C_DIM 512
#define HW    4096
#define IP    10

typedef __attribute__((ext_vector_type(16))) __bf16        v16bf;
typedef __attribute__((ext_vector_type(8)))  float         v8f;
typedef __attribute__((ext_vector_type(4)))  unsigned int  u32x4;

union Frag {            // 16 bf16 values per lane = 8 VGPRs
    v16bf v;
    u32x4 q[2];
};

// Upper-triangular 128x128 tile pairs of the symmetric 512x512 Gram matrix.
__constant__ int kTM[10] = {0, 0, 0, 0, 1, 1, 1, 2, 2, 3};
__constant__ int kTN[10] = {0, 1, 2, 3, 1, 2, 3, 2, 3, 3};

__device__ __forceinline__ unsigned short f32_to_bf16_rne(float f) {
    unsigned int u = __float_as_uint(f);
    u += 0x7FFFu + ((u >> 16) & 1u);
    return (unsigned short)(u >> 16);
}

// ---------------------------------------------------------------------------
// 1) Convert x (fp32) -> bf16 workspace copy; zero the ||u||^2 accumulators.
// ---------------------------------------------------------------------------
__global__ __launch_bounds__(256)
void convert_kernel(const float* __restrict__ x,
                    unsigned short* __restrict__ xbf,
                    float* __restrict__ unorm)
{
    size_t i4 = ((size_t)blockIdx.x * 256 + threadIdx.x) * 4;
    float4 f = *(const float4*)(x + i4);
    ushort4 o;
    o.x = f32_to_bf16_rne(f.x);
    o.y = f32_to_bf16_rne(f.y);
    o.z = f32_to_bf16_rne(f.z);
    o.w = f32_to_bf16_rne(f.w);
    *(ushort4*)(xbf + i4) = o;
    if (blockIdx.x == 0 && threadIdx.x < B_DIM) unorm[threadIdx.x] = 0.0f;
}

// ---------------------------------------------------------------------------
// 2) Gram matrix Ws[b] = X[b] * X[b]^T via v_wmma_f32_16x16x32_bf16.
//    Symmetric: only the 10 upper-triangular 128x128 tiles are computed;
//    off-diagonal tiles are mirrored at store time.
//    Block = 256 threads = 8 waves as 2 (M) x 4 (N); block tile = 128x128.
//    Wave tile = 64x32 = 4x2 WMMA accumulators; K loop 4096 in steps of 32.
//
//    Fragment layouts per CDNA5 ISA (wave32):
//      A (16x32 bf16): lane = half*16+m holds row m; VGPR p holds
//        K = (p<4?0:16) + half*8 + 2*(p&3) + {0,1}  -> two 16B row chunks.
//      B (32x16 bf16): lane = half*16+m holds col n=m; element e holds
//        K = (e<8?0:16) + half*8 + (e&7)           -> two 16B row chunks, same as A.
//      C/D (16x16 f32): VGPR r -> row half*8+r, col m.
// ---------------------------------------------------------------------------
__global__ __launch_bounds__(256)
void gram_kernel(const unsigned short* __restrict__ xbf,
                 float* __restrict__ Ws)
{
    const int b    = blockIdx.z;
    const int tM   = kTM[blockIdx.x];     // 0..3 (uniform per block)
    const int tN   = kTN[blockIdx.x];     // 0..3, tN >= tM
    const int tid  = threadIdx.x;
    const int wave = tid >> 5;
    const int lane = tid & 31;
    const int half = lane >> 4;           // 0/1
    const int m    = lane & 15;           // 0..15
    const int waveM = wave >> 2;          // 0..1
    const int waveN = wave & 3;           // 0..3

    const int i0 = tM * 128 + waveM * 64; // wave M origin (+ mi*16)
    const int j0 = tN * 128 + waveN * 32; // wave N origin (+ ni*16)

    const unsigned short* xb = xbf + (size_t)b * C_DIM * HW;

    // Per-lane row pointers (A rows carry matrix row m; B rows carry col n=m).
    const unsigned short* Arow[4];
    const unsigned short* Brow[2];
#pragma unroll
    for (int mi = 0; mi < 4; ++mi)
        Arow[mi] = xb + (size_t)(i0 + mi * 16 + m) * HW + half * 8;
#pragma unroll
    for (int ni = 0; ni < 2; ++ni)
        Brow[ni] = xb + (size_t)(j0 + ni * 16 + m) * HW + half * 8;

    v8f acc[4][2];
#pragma unroll
    for (int mi = 0; mi < 4; ++mi)
#pragma unroll
        for (int ni = 0; ni < 2; ++ni)
            acc[mi][ni] = (v8f){0.f, 0.f, 0.f, 0.f, 0.f, 0.f, 0.f, 0.f};

    for (int k0 = 0; k0 < HW; k0 += 32) {
        Frag A[4], Bf[2];
#pragma unroll
        for (int mi = 0; mi < 4; ++mi) {
            A[mi].q[0] = *(const u32x4*)(Arow[mi] + k0);       // K chunk 0..15
            A[mi].q[1] = *(const u32x4*)(Arow[mi] + k0 + 16);  // K chunk 16..31
        }
#pragma unroll
        for (int ni = 0; ni < 2; ++ni) {
            const u32x4* p = (const u32x4*)(Brow[ni] + k0);    // two 16B chunks, same K as A
            Bf[ni].q[0] = p[0];
            Bf[ni].q[1] = p[2];
        }
#pragma unroll
        for (int mi = 0; mi < 4; ++mi)
#pragma unroll
            for (int ni = 0; ni < 2; ++ni)
                acc[mi][ni] = __builtin_amdgcn_wmma_f32_16x16x32_bf16(
                    /*neg_a=*/false, A[mi].v,
                    /*neg_b=*/false, Bf[ni].v,
                    /*c_mod=*/(short)0, acc[mi][ni],
                    /*reuse_a=*/false, /*reuse_b=*/false);
    }

    float* Wb = Ws + (size_t)b * C_DIM * C_DIM;
    const bool mirror = (tM != tN);
#pragma unroll
    for (int mi = 0; mi < 4; ++mi) {
#pragma unroll
        for (int ni = 0; ni < 2; ++ni) {
            const int col = j0 + ni * 16 + m;
#pragma unroll
            for (int r = 0; r < 8; ++r) {
                const int row = i0 + mi * 16 + half * 8 + r;
                const float val = acc[mi][ni][r];
                Wb[(size_t)row * C_DIM + col] = val;
                if (mirror) Wb[(size_t)col * C_DIM + row] = val;
            }
        }
    }
}

// ---------------------------------------------------------------------------
// 3) 10 power-iteration steps: v = normalize(Ws v), one block per batch.
// ---------------------------------------------------------------------------
__global__ __launch_bounds__(512)
void power_iter_kernel(const float* __restrict__ Ws,
                       const float* __restrict__ v_in,
                       float* __restrict__ v_out)
{
    __shared__ float vL[C_DIM];
    __shared__ float red[C_DIM];
    const int b   = blockIdx.x;
    const int tid = threadIdx.x;
    const float* Wb = Ws + (size_t)b * C_DIM * C_DIM;

    vL[tid] = v_in[b * C_DIM + tid];
    __syncthreads();

    float y = 0.0f;
    for (int it = 0; it < IP; ++it) {
        const float* row = Wb + (size_t)tid * C_DIM;
        y = 0.0f;
#pragma unroll 8
        for (int j = 0; j < C_DIM; ++j) y += row[j] * vL[j];
        red[tid] = y * y;
        __syncthreads();
        for (int s = 256; s > 0; s >>= 1) {
            if (tid < s) red[tid] += red[tid + s];
            __syncthreads();
        }
        const float inv = rsqrtf(red[0]);
        vL[tid] = y * inv;
        __syncthreads();
    }
    v_out[b * C_DIM + tid] = vL[tid];
}

// ---------------------------------------------------------------------------
// 4) u[b,n] = sum_c x[b,c,n] * v[b,c]; accumulate ||u||^2 per batch.
//    Each thread handles 4 consecutive n (float4 streaming loads).
// ---------------------------------------------------------------------------
__global__ __launch_bounds__(256)
void u_kernel(const float* __restrict__ x,
              const float* __restrict__ v,
              float* __restrict__ u,
              float* __restrict__ unorm)
{
    __shared__ float vsh[C_DIM];
    __shared__ float red[256];
    const int b   = blockIdx.y;
    const int tid = threadIdx.x;
    for (int j = tid; j < C_DIM; j += 256) vsh[j] = v[b * C_DIM + j];
    __syncthreads();

    const int n0 = (blockIdx.x * 256 + tid) * 4;
    const float* xb = x + (size_t)b * C_DIM * HW + n0;
    float4 acc = make_float4(0.f, 0.f, 0.f, 0.f);
#pragma unroll 4
    for (int c = 0; c < C_DIM; ++c) {
        const float4 xf = *(const float4*)(xb + (size_t)c * HW);
        const float vc = vsh[c];
        acc.x += xf.x * vc;
        acc.y += xf.y * vc;
        acc.z += xf.z * vc;
        acc.w += xf.w * vc;
    }
    *(float4*)(u + b * HW + n0) = acc;

    red[tid] = acc.x * acc.x + acc.y * acc.y + acc.z * acc.z + acc.w * acc.w;
    __syncthreads();
    for (int s = 128; s > 0; s >>= 1) {
        if (tid < s) red[tid] += red[tid + s];
        __syncthreads();
    }
    if (tid == 0) atomicAdd(&unorm[b], red[0]);
}

// ---------------------------------------------------------------------------
// 5) out[b,c,n] = x[b,c,n] + (u[b,n]/||u||) * v[b,c]   (float4 per thread)
// ---------------------------------------------------------------------------
__global__ __launch_bounds__(256)
void final_kernel(const float* __restrict__ x,
                  const float* __restrict__ u,
                  const float* __restrict__ v,
                  const float* __restrict__ unorm,
                  float* __restrict__ out)
{
    const size_t i4 = ((size_t)blockIdx.x * 256 + threadIdx.x) * 4;
    const int b = (int)(i4 >> 21);              // C*HW = 2^21
    const int c = (int)((i4 >> 12) & (C_DIM - 1));
    const int n = (int)(i4 & (HW - 1));         // 4-aligned
    const float s = rsqrtf(unorm[b]) * v[b * C_DIM + c];
    const float4 xf = *(const float4*)(x + i4);
    const float4 uf = *(const float4*)(u + b * HW + n);
    float4 o;
    o.x = xf.x + uf.x * s;
    o.y = xf.y + uf.y * s;
    o.z = xf.z + uf.z * s;
    o.w = xf.w + uf.w * s;
    *(float4*)(out + i4) = o;
}

// ---------------------------------------------------------------------------
extern "C" void kernel_launch(void* const* d_in, const int* in_sizes, int n_in,
                              void* d_out, int out_size, void* d_ws, size_t ws_size,
                              hipStream_t stream)
{
    (void)in_sizes; (void)n_in; (void)out_size; (void)ws_size;
    const float* x    = (const float*)d_in[0];   // (B, C, H, W) fp32
    const float* v_in = (const float*)d_in[1];   // (B, C, 1)   fp32
    float* out = (float*)d_out;

    // Workspace layout
    char* ws = (char*)d_ws;
    unsigned short* xbf = (unsigned short*)ws;                          // 64 MiB
    float* Ws    = (float*)(ws + (size_t)B_DIM * C_DIM * HW * 2);       // 16 MiB
    float* v_cur = (float*)((char*)Ws + (size_t)B_DIM * C_DIM * C_DIM * 4);
    float* u     = v_cur + (size_t)B_DIM * C_DIM;
    float* unorm = u + (size_t)B_DIM * HW;

    const size_t total = (size_t)B_DIM * C_DIM * HW;                    // 33,554,432

    convert_kernel<<<(unsigned)(total / (256 * 4)), 256, 0, stream>>>(x, xbf, unorm);

    dim3 ggrid(10, 1, B_DIM);   // 10 upper-triangular 128x128 tiles per batch
    gram_kernel<<<ggrid, 256, 0, stream>>>(xbf, Ws);

    power_iter_kernel<<<B_DIM, C_DIM, 0, stream>>>(Ws, v_in, v_cur);

    dim3 ugrid(HW / (256 * 4), B_DIM);
    u_kernel<<<ugrid, 256, 0, stream>>>(x, v_cur, u, unorm);

    final_kernel<<<(unsigned)(total / (256 * 4)), 256, 0, stream>>>(x, u, v_cur, unorm, out);
}
